// DiscreteAutoregressiveFlow_85186381349467
// MI455X (gfx1250) — matrix operand order repaired and checked by reading, not compiled, not run on hardware
//
#include <hip/hip_runtime.h>
#include <hip/hip_bf16.h>
#include <stdint.h>

#define V 257
#define L 1024
#define NSTATE 258   // states 0..256 plus sentinel (zero vector) at 257

typedef __attribute__((ext_vector_type(16))) _Float16 v16h;
typedef __attribute__((ext_vector_type(8)))  float    v8f;

// ---------------------------------------------------------------------------
// Kernel 1: zero the 67MB output (float4 grid-stride stores)
// ---------------------------------------------------------------------------
__global__ void daf_zero_kernel(float4* __restrict__ out, int n4) {
    int i = blockIdx.x * blockDim.x + threadIdx.x;
    int stride = gridDim.x * blockDim.x;
    float4 z = make_float4(0.f, 0.f, 0.f, 0.f);
    for (; i < n4; i += stride) out[i] = z;
}

// ---------------------------------------------------------------------------
// Kernel 2: per-state tables. For each state p (row of W, or b-only sentinel):
//   tblL[p] = argmax_j (W[p,j]+b[j])                 j in [0,V)
//   tblA[p] = inv_table[ argmax_j (W[p,V+j]+b[V+j]) ]
// First-max tie-break (JAX argmax semantics) via ordered-uint key | (1024-j).
// ---------------------------------------------------------------------------
__device__ __forceinline__ unsigned ordered_key(float v) {
    unsigned fb = __float_as_uint(v);
    return (fb & 0x80000000u) ? ~fb : (fb | 0x80000000u);
}

__global__ void daf_tables_kernel(const float* __restrict__ W,
                                  const float* __restrict__ bias,
                                  const int*   __restrict__ inv_table,
                                  int* __restrict__ tblL,
                                  int* __restrict__ tblA) {
    __shared__ unsigned long long rl[256];
    __shared__ unsigned long long rs[256];
    int p = blockIdx.x;                 // 0..257
    int t = threadIdx.x;                // 0..255
    const float* row = (p < V) ? (W + (size_t)p * (2 * V)) : nullptr;

    unsigned long long bestL = 0, bestS = 0;
    for (int j = t; j < V; j += 256) {
        float vl = bias[j]     + (row ? row[j]     : 0.f);
        float vs = bias[V + j] + (row ? row[V + j] : 0.f);
        unsigned long long kl = ((unsigned long long)ordered_key(vl) << 32) | (unsigned)(1024 - j);
        unsigned long long ks = ((unsigned long long)ordered_key(vs) << 32) | (unsigned)(1024 - j);
        if (kl > bestL) bestL = kl;
        if (ks > bestS) bestS = ks;
    }
    rl[t] = bestL; rs[t] = bestS;
    __syncthreads();
    for (int off = 128; off > 0; off >>= 1) {
        if (t < off) {
            if (rl[t + off] > rl[t]) rl[t] = rl[t + off];
            if (rs[t + off] > rs[t]) rs[t] = rs[t + off];
        }
        __syncthreads();
    }
    if (t == 0) {
        int l = 1024 - (int)(rl[0] & 0xFFFFFFFFull);
        int s = 1024 - (int)(rs[0] & 0xFFFFFFFFull);
        tblL[p] = l;
        tblA[p] = inv_table[s];
    }
}

// ---------------------------------------------------------------------------
// Kernel 3: recover input indices vIdx[row] = sum_k x[row,k]*k  (exact: x is
// a 0/1 one-hot; indices <= 256 are exact in f16 products, f32 accumulate).
// One wave per 16 rows; K accumulated via v_wmma_f32_16x16x32_f16 in 8 chunks
// (K=0..255), column 256 added scalarly. B matrix = iota replicated over N.
//   A (16-bit 16x32): lane = half*16+m, halves 0..7 -> K=half*8+j,
//                     halves 8..15 -> K=half*8+16+(h-8).
//   B (16-bit 32x16): same K order as A (half*8+j, then half*8+16+j).
//   D (f32 16x16):    lane n (0..15 / 16..31), VGPR r -> M=r / M=8+r.
// ---------------------------------------------------------------------------
__global__ void daf_vidx_wmma_kernel(const float* __restrict__ X,
                                     int* __restrict__ vIdx, int nRows) {
    int gtid = blockIdx.x * blockDim.x + threadIdx.x;
    int wave = gtid >> 5;
    int lane = threadIdx.x & 31;
    int rowBase = wave * 16;
    if (rowBase >= nRows) return;            // wave-uniform: EXEC stays all-1s

    int half = lane >> 4;                    // 0 or 1
    int m    = lane & 15;                    // row within 16-row tile
    const float* arow = X + (size_t)(rowBase + m) * V;
    int kbaseA = half * 8;
    int kbaseB = half * 8;

    v8f acc = {0.f, 0.f, 0.f, 0.f, 0.f, 0.f, 0.f, 0.f};
    for (int c0 = 0; c0 < 256; c0 += 32) {
        v16h a, bf;
        #pragma unroll
        for (int j = 0; j < 8; ++j) {
            a[j]     = (_Float16)arow[c0 + kbaseA + j];
            a[8 + j] = (_Float16)arow[c0 + kbaseA + 16 + j];
        }
        #pragma unroll
        for (int h = 0; h < 16; ++h) {
            bf[h] = (_Float16)(float)(c0 + kbaseB + h + (h >= 8 ? 8 : 0));
        }
        acc = __builtin_amdgcn_wmma_f32_16x16x32_f16(
            /*neg_a=*/false, a, /*neg_b=*/false, bf,
            /*c_mod=*/(short)0, acc, /*reuse_a=*/false, /*reuse_b=*/false);
    }
    if (m == 0) {                            // lanes 0 (M=0..7) and 16 (M=8..15)
        int mb = rowBase + half * 8;
        #pragma unroll
        for (int r = 0; r < 8; ++r) {
            int row = mb + r;
            float val = acc[r] + X[(size_t)row * V + (V - 1)] * 256.0f;
            vIdx[row] = (int)(val + 0.5f);
        }
    }
}

// ---------------------------------------------------------------------------
// Kernel 4: the autoregressive scan, reduced to an integer recurrence.
// One lane per batch chain; tables live in LDS (one ds_load pair per step).
//   p = 257 (sentinel)              initially
//   l = tblL[p]; a = tblA[p]
//   a==0 -> zero row, p := sentinel
//   else u = ((v - l + V) * a) % V ; out[b,t,u] = 1 ; p := u
// ---------------------------------------------------------------------------
__global__ void daf_scan_kernel(const int* __restrict__ vIdx,
                                const int* __restrict__ tblL,
                                const int* __restrict__ tblA,
                                float* __restrict__ out, int Bn) {
    __shared__ int sL[NSTATE];
    __shared__ int sA[NSTATE];
    for (int i = threadIdx.x; i < NSTATE; i += blockDim.x) {
        sL[i] = tblL[i];
        sA[i] = tblA[i];
    }
    __syncthreads();

    int b = blockIdx.x * blockDim.x + threadIdx.x;
    if (b >= Bn) return;

    const int* vrow = vIdx + (size_t)b * L;
    float* orow = out + (size_t)b * L * V;

    int p = V;                                // sentinel state
    for (int t = 0; t < L; ++t) {
        __builtin_prefetch(&vrow[t + 16], 0, 0);   // global_prefetch_b8
        int v = vrow[t];
        int l = sL[p];
        int a = sA[p];
        if (a == 0) {
            p = V;                            // zero output row (already zeroed)
        } else {
            unsigned u = ((unsigned)(v - l + V) * (unsigned)a) % (unsigned)V;
            orow[(size_t)t * V + u] = 1.0f;
            p = (int)u;
        }
    }
}

// ---------------------------------------------------------------------------
// Launcher
// ---------------------------------------------------------------------------
extern "C" void kernel_launch(void* const* d_in, const int* in_sizes, int n_in,
                              void* d_out, int out_size, void* d_ws, size_t ws_size,
                              hipStream_t stream) {
    const float* x         = (const float*)d_in[0];   // [B, L, V]
    const float* W         = (const float*)d_in[1];   // [V, 2V]
    const float* bias      = (const float*)d_in[2];   // [2V]
    const int*   inv_table = (const int*)  d_in[3];   // [V]
    float* out = (float*)d_out;

    const int Bn    = in_sizes[0] / (L * V);          // 64
    const int nRows = Bn * L;                         // 65536

    // workspace layout
    int* vIdx = (int*)d_ws;                           // nRows ints
    int* tblL = vIdx + nRows;                         // NSTATE ints
    int* tblA = tblL + NSTATE;                        // NSTATE ints

    // 1) zero output
    int n4 = out_size / 4;
    daf_zero_kernel<<<4096, 256, 0, stream>>>((float4*)out, n4);

    // 2) per-state argmax tables (258 states)
    daf_tables_kernel<<<NSTATE, 256, 0, stream>>>(W, bias, inv_table, tblL, tblA);

    // 3) one-hot -> index via WMMA (128 rows per 256-thread block)
    int vblocks = (nRows + 127) / 128;
    daf_vidx_wmma_kernel<<<vblocks, 256, 0, stream>>>(x, vIdx, nRows);

    // 4) integer scan, one lane per batch chain
    daf_scan_kernel<<<1, 128, 0, stream>>>(vIdx, tblL, tblA, out, Bn);
}
